// QuantumMLP_80736795230972
// MI455X (gfx1250) — matrix operand order repaired and checked by reading, not compiled, not run on hardware
//
#include <hip/hip_runtime.h>
#include <hip/hip_bf16.h>

typedef __attribute__((ext_vector_type(16))) _Float16 v16h;
typedef __attribute__((ext_vector_type(8)))  _Float16 v8h;
typedef __attribute__((ext_vector_type(8)))  float    v8f;

#define N_QUBITS 8
#define DEPTH    2
#define BATCH    8192
#define IN_F     256
#define OUT_F    128
#define QDIM     256   // 2^8

static __device__ __forceinline__ float shfx(float v, int mask) {
    return __shfl_xor(v, mask, 32);
}

// RY on bit position `bit` of the 8-bit amp index. amp = (lane<<3)|r.
// new0 = c*a0 - s*a1 ; new1 = s*a0 + c*a1  (applies identically to re & im)
static __device__ __forceinline__ void apply_ry(float re[8], float im[8],
                                                int bit, float c, float s, int lane) {
    if (bit < 3) {
        const int m = 1 << bit;
#pragma unroll
        for (int r = 0; r < 8; ++r) {
            if (!(r & m)) {
                const int r1 = r | m;
                float a0r = re[r], a0i = im[r], a1r = re[r1], a1i = im[r1];
                re[r]  = c * a0r - s * a1r;  im[r]  = c * a0i - s * a1i;
                re[r1] = s * a0r + c * a1r;  im[r1] = s * a0i + c * a1i;
            }
        }
    } else {
        const int lm = 1 << (bit - 3);
        const float sg = (lane & lm) ? s : -s;   // bit=0: c*mine - s*other ; bit=1: c*mine + s*other
#pragma unroll
        for (int r = 0; r < 8; ++r) {
            float orr = shfx(re[r], lm);
            float oii = shfx(im[r], lm);
            re[r] = c * re[r] + sg * orr;
            im[r] = c * im[r] + sg * oii;
        }
    }
}

// RX: new0 = c*a0 - i*s*a1 ; new1 = c*a1 - i*s*a0
// => re' = c*re_mine + s*im_other ; im' = c*im_mine - s*re_other (symmetric in halves)
static __device__ __forceinline__ void apply_rx(float re[8], float im[8],
                                                int bit, float c, float s, int lane) {
    if (bit < 3) {
        const int m = 1 << bit;
#pragma unroll
        for (int r = 0; r < 8; ++r) {
            if (!(r & m)) {
                const int r1 = r | m;
                float r0 = re[r], i0 = im[r], r1v = re[r1], i1v = im[r1];
                re[r]  = c * r0  + s * i1v;  im[r]  = c * i0  - s * r1v;
                re[r1] = c * r1v + s * i0;   im[r1] = c * i1v - s * r0;
            }
        }
    } else {
        const int lm = 1 << (bit - 3);
#pragma unroll
        for (int r = 0; r < 8; ++r) {
            float orr = shfx(re[r], lm);
            float oii = shfx(im[r], lm);
            re[r] = c * re[r] + s * oii;
            im[r] = c * im[r] - s * orr;
        }
    }
}

// CNOT: swap amps with ctrl-bit==1 across the target bit.
static __device__ __forceinline__ void apply_cnot(float re[8], float im[8],
                                                  int bc, int bt, int lane) {
    if (bc < 3 && bt < 3) {                       // both in-register
        const int mc = 1 << bc, mt = 1 << bt;
#pragma unroll
        for (int r = 0; r < 8; ++r) {
            if ((r & mc) && !(r & mt)) {
                const int r1 = r | mt;
                float t;
                t = re[r]; re[r] = re[r1]; re[r1] = t;
                t = im[r]; im[r] = im[r1]; im[r1] = t;
            }
        }
    } else if (bc >= 3 && bt < 3) {               // ctrl on lane bit, tgt in-register
        const int mt = 1 << bt;
        if (lane & (1 << (bc - 3))) {
#pragma unroll
            for (int r = 0; r < 8; ++r) {
                if (!(r & mt)) {
                    const int r1 = r | mt;
                    float t;
                    t = re[r]; re[r] = re[r1]; re[r1] = t;
                    t = im[r]; im[r] = im[r1]; im[r1] = t;
                }
            }
        }
    } else if (bc < 3 && bt >= 3) {               // ctrl in-register, tgt cross-lane
        const int mc = 1 << bc, lm = 1 << (bt - 3);
#pragma unroll
        for (int r = 0; r < 8; ++r) {
            if (r & mc) {                          // uniform across lanes: no divergence
                re[r] = shfx(re[r], lm);
                im[r] = shfx(im[r], lm);
            }
        }
    } else {                                       // both cross-lane
        const int lmc = 1 << (bc - 3), lmt = 1 << (bt - 3);
        const bool act = (lane & lmc) != 0;        // partner lane has same ctrl bit
#pragma unroll
        for (int r = 0; r < 8; ++r) {
            float orr = shfx(re[r], lmt);
            float oii = shfx(im[r], lmt);
            if (act) { re[r] = orr; im[r] = oii; }
        }
    }
}

// ---------------- kernel 1: W_out f32 -> f16 ----------------
__global__ void qmlp_convert_w(const float* __restrict__ w, _Float16* __restrict__ w16) {
    int i = blockIdx.x * blockDim.x + threadIdx.x;   // OUT_F*QDIM = 32768 total
    w16[i] = (_Float16)w[i];
}

// ---------------- kernel 2: fused angles + 8-qubit circuit -> probs (f16) ----------------
// One sample per wave32; 8 waves per block.
__global__ void __launch_bounds__(256)
qmlp_circuit(const float* __restrict__ x, const float* __restrict__ W_in,
             const float* __restrict__ b_in, const float* __restrict__ qw,
             const float* __restrict__ us, _Float16* __restrict__ probs16) {
    const int lane = threadIdx.x & 31;
    const int s    = blockIdx.x * 8 + (threadIdx.x >> 5);

    // angles[i] = tanh(x[s]·W_in[i] + b_in[i]) * pi   (wave-cooperative dot products)
    float ang[N_QUBITS];
#pragma unroll
    for (int i = 0; i < N_QUBITS; ++i) {
        float acc = 0.f;
        const float* xr = x + (size_t)s * IN_F;
        const float* wr = W_in + (size_t)i * IN_F;
#pragma unroll
        for (int k = 0; k < IN_F / 32; ++k)
            acc += xr[lane + 32 * k] * wr[lane + 32 * k];
#pragma unroll
        for (int m = 16; m >= 1; m >>= 1) acc += shfx(acc, m);
        ang[i] = tanhf(acc + b_in[i]) * 3.14159265358979f;
    }

    // |0...0> state: amp index = (lane<<3)|r ; qubit i <-> bit (7-i)
    float re[8], im[8];
#pragma unroll
    for (int r = 0; r < 8; ++r) { re[r] = 0.f; im[r] = 0.f; }
    if (lane == 0) re[0] = 1.0f;

    // initial data-encoding RY layer
#pragma unroll
    for (int i = 0; i < N_QUBITS; ++i) {
        float sn, cs; __sincosf(0.5f * ang[i], &sn, &cs);
        apply_ry(re, im, 7 - i, cs, sn, lane);
    }

#pragma unroll
    for (int l = 0; l < DEPTH; ++l) {
#pragma unroll
        for (int i = 0; i < N_QUBITS; ++i) {            // re-upload RY
            float th = ang[i] * us[l * N_QUBITS + i];
            float sn, cs; __sincosf(0.5f * th, &sn, &cs);
            apply_ry(re, im, 7 - i, cs, sn, lane);
        }
#pragma unroll
        for (int i = 0; i < N_QUBITS; ++i) {            // RX, RY trainables
            float sn, cs;
            __sincosf(0.5f * qw[(l * N_QUBITS + i) * 3 + 0], &sn, &cs);
            apply_rx(re, im, 7 - i, cs, sn, lane);
            __sincosf(0.5f * qw[(l * N_QUBITS + i) * 3 + 1], &sn, &cs);
            apply_ry(re, im, 7 - i, cs, sn, lane);
        }
#pragma unroll
        for (int i = 0; i < N_QUBITS; ++i)              // CNOT ring
            apply_cnot(re, im, 7 - i, 7 - ((i + 1) & 7), lane);
#pragma unroll
        for (int i = 0; i < N_QUBITS; ++i) {            // final RY of layer
            float sn, cs;
            __sincosf(0.5f * qw[(l * N_QUBITS + i) * 3 + 2], &sn, &cs);
            apply_ry(re, im, 7 - i, cs, sn, lane);
        }
    }

    // probs -> f16, vectorized 16B store per lane
    v8h pv;
#pragma unroll
    for (int r = 0; r < 8; ++r)
        pv[r] = (_Float16)(re[r] * re[r] + im[r] * im[r]);
    *(v8h*)(probs16 + (size_t)s * QDIM + lane * 8) = pv;
}

// ---------------- kernel 3: out = probs @ W_out^T + b_out  via WMMA f16->f32 ----------------
// One 16x16 output tile per wave; K-loop 256 in steps of 32 -> 8 v_wmma per wave.
__global__ void __launch_bounds__(256)
qmlp_gemm(const _Float16* __restrict__ probs, const _Float16* __restrict__ w16,
          const float* __restrict__ b_out, float* __restrict__ out) {
    const int lane = threadIdx.x & 31;
    const int w    = blockIdx.x * 8 + (threadIdx.x >> 5);
    const int mt   = w >> 3;            // 512 M-tiles
    const int nt   = w & 7;             // 8 N-tiles
    const int row_base = mt * 16;
    const int col_base = nt * 16;
    const int m = lane & 15;            // A row / B col / C col for this lane
    const int h = lane >> 4;            // half-wave

    // C/D layout: c[v] = D[row_base + v + 8*h][col_base + m]
    v8f c;
    const float bias = b_out[col_base + m];
#pragma unroll
    for (int v = 0; v < 8; ++v) c[v] = bias;

    const _Float16* arow = probs + (size_t)(row_base + m) * QDIM;
    const _Float16* brow = w16   + (size_t)(col_base + m) * QDIM;

#pragma unroll
    for (int k0 = 0; k0 < QDIM; k0 += 32) {
        // A 16x32 f16: lane m, half h: j<8 -> K = k0 + h*8 + j ; j>=8 -> K = k0 + 16 + h*8 + (j-8)
        v8h alo = *(const v8h*)(arow + k0 + h * 8);
        v8h ahi = *(const v8h*)(arow + k0 + 16 + h * 8);
        v16h a, b;
#pragma unroll
        for (int j = 0; j < 8; ++j) { a[j] = alo[j]; a[j + 8] = ahi[j]; }
        // B 32x16 f16: lane col m, half h: j<8 -> K = k0 + h*8 + j ; j>=8 -> K = k0 + 16 + h*8 + (j-8)
        v8h blo = *(const v8h*)(brow + k0 + h * 8);
        v8h bhi = *(const v8h*)(brow + k0 + 16 + h * 8);
#pragma unroll
        for (int j = 0; j < 8; ++j) { b[j] = blo[j]; b[j + 8] = bhi[j]; }
        c = __builtin_amdgcn_wmma_f32_16x16x32_f16(false, a, false, b, (short)0, c, false, false);
    }

#pragma unroll
    for (int v = 0; v < 8; ++v)
        out[(size_t)(row_base + v + 8 * h) * OUT_F + col_base + m] = c[v];
}

extern "C" void kernel_launch(void* const* d_in, const int* in_sizes, int n_in,
                              void* d_out, int out_size, void* d_ws, size_t ws_size,
                              hipStream_t stream) {
    const float* x     = (const float*)d_in[0];   // (8192, 256)
    const float* W_in  = (const float*)d_in[1];   // (8, 256)
    const float* b_in  = (const float*)d_in[2];   // (8,)
    const float* qw    = (const float*)d_in[3];   // (2, 8, 3)
    const float* us    = (const float*)d_in[4];   // (2, 8)
    const float* W_out = (const float*)d_in[5];   // (128, 256)
    const float* b_out = (const float*)d_in[6];   // (128,)
    float* out = (float*)d_out;                   // (8192, 128)

    // workspace layout: [0, 4MiB) probs f16 ; [4MiB, 4MiB+64KiB) W_out f16
    _Float16* probs16 = (_Float16*)d_ws;
    _Float16* w16     = (_Float16*)((char*)d_ws + (size_t)BATCH * QDIM * sizeof(_Float16));

    qmlp_convert_w<<<(OUT_F * QDIM) / 256, 256, 0, stream>>>(W_out, w16);
    qmlp_circuit<<<BATCH / 8, 256, 0, stream>>>(x, W_in, b_in, qw, us, probs16);
    qmlp_gemm<<<(BATCH / 16) * (OUT_F / 16) / 8, 256, 0, stream>>>(probs16, w16, b_out, out);
}
